// LiteMLA_61297773248523
// MI455X (gfx1250) — matrix operand order repaired and checked by reading, not compiled, not run on hardware
//
#include <hip/hip_runtime.h>
#include <hip/hip_bf16.h>

// ---------------------------------------------------------------------------
// LiteMLA forward on gfx1250 (MI455X): bf16 WMMA everywhere, f32 accumulate.
//   B=4, N=4096, C=1024, D=32, h=32.  EPS=1e-15, PAD_VAL=1.0
// GEMMs: 128x256 block tile, 8 waves as 2x4, 64x64 wave tile (4x4 frags),
// BK=32, double-buffered LDS (one barrier per K-step). All fragments are
// loaded into distinct registers ahead of the WMMA burst so ds_loads pipeline
// (partial s_wait_dscnt) instead of serializing per 4-WMMA group.
// ---------------------------------------------------------------------------

typedef __attribute__((ext_vector_type(16))) __bf16 v16bf;
typedef __attribute__((ext_vector_type(8)))  float  v8f;

#define WMMA_BF16(a, b, c) \
  __builtin_amdgcn_wmma_f32_16x16x32_bf16(false, (a), false, (b), (short)0, (c), false, false)

union Frag  { uint4 u[2]; v16bf v; };   // 16 bf16 per lane (A or B fragment)
union Pack8 { uint4 u; __bf16 h[8]; };  // 8 bf16 = 16B vector store
union Pack4 { uint2 u; __bf16 h[4]; };

__device__ __forceinline__ v8f v8f_zero() {
  v8f z;
#pragma unroll
  for (int i = 0; i < 8; ++i) z[i] = 0.0f;
  return z;
}

// ------------------------------- conversions -------------------------------
__global__ __launch_bounds__(256) void cvt_f32_bf16(const float* __restrict__ in,
                                                    __bf16* __restrict__ out, int n) {
  int i = (blockIdx.x * 256 + threadIdx.x) * 4;
  if (i + 3 < n) {
    float4 f = *(const float4*)(in + i);
    Pack4 p;
    p.h[0] = (__bf16)f.x; p.h[1] = (__bf16)f.y;
    p.h[2] = (__bf16)f.z; p.h[3] = (__bf16)f.w;
    *(uint2*)(out + i) = p.u;
  } else {
    for (; i < n; ++i) out[i] = (__bf16)in[i];
  }
}

// vbuf rows 32..47 per (b,h): row 32 = 1.0 (PAD), rows 33..47 = 0 (tile pad)
__global__ __launch_bounds__(256) void init_vpad(__bf16* __restrict__ vbuf) {
  int i = blockIdx.x * 256 + threadIdx.x;      // 128 * 16 * 4096 threads
  if (i >= 128 * 16 * 4096) return;
  int n  = i & 4095;
  int r  = (i >> 12) & 15;
  int bh = i >> 16;
  vbuf[((size_t)bh * 48 + 32 + r) * 4096 + n] = (__bf16)((r == 0) ? 1.0f : 0.0f);
}

// ------------------------- shared WMMA GEMM mainloop ------------------------
// Y[M,Ntok] = A[M,K] * B[Ntok,K]^T   (both row-major, K contiguous), K=1024.
#define LDT 40  // LDS row stride in bf16 elems (pad to dodge bank conflicts)
#define A_BUF (128 * LDT)
#define B_BUF (256 * LDT)

__device__ __forceinline__ void stage_tiles(const __bf16* __restrict__ Ag,
                                            const __bf16* __restrict__ Bg,
                                            int K, int k0,
                                            __bf16* sA, __bf16* sB) {
  const int tid = threadIdx.x;
  {  // A: 128 rows x 32 cols, 2 x b128 per thread
    const int r = tid >> 1, c = (tid & 1) * 16;
    const __bf16* ap = Ag + (size_t)r * K + k0 + c;
    *(uint4*)(sA + r * LDT + c)     = *(const uint4*)(ap);
    *(uint4*)(sA + r * LDT + c + 8) = *(const uint4*)(ap + 8);
  }
  {  // B: 256 rows x 32 cols, 4 x b128 per thread (one full row)
    const int r = tid;
    const __bf16* bp = Bg + (size_t)r * K + k0;
    *(uint4*)(sB + r * LDT + 0)  = *(const uint4*)(bp);
    *(uint4*)(sB + r * LDT + 8)  = *(const uint4*)(bp + 8);
    *(uint4*)(sB + r * LDT + 16) = *(const uint4*)(bp + 16);
    *(uint4*)(sB + r * LDT + 24) = *(const uint4*)(bp + 24);
  }
}

__device__ __forceinline__ void gemm_mainloop(const __bf16* __restrict__ Ag,
                                              const __bf16* __restrict__ Bg,
                                              int K, __bf16* sA, __bf16* sB,
                                              v8f (&acc)[4][4]) {
  const int tid  = threadIdx.x;
  const int lane = tid & 31, wave = tid >> 5;
  const int wm = wave >> 2, wn = wave & 3;
  const int l15 = lane & 15, hi = lane >> 4;

  stage_tiles(Ag, Bg, K, 0, sA, sB);     // prologue fill of buffer 0
  for (int k0 = 0; k0 < K; k0 += 32) {
    const int cur = (k0 >> 5) & 1;
    const __bf16* cA = sA + cur * A_BUF;
    const __bf16* cB = sB + cur * B_BUF;
    __syncthreads();                     // buffer `cur` ready; `cur^1` free
    if (k0 + 32 < K)
      stage_tiles(Ag, Bg, K, k0 + 32, sA + (cur ^ 1) * A_BUF, sB + (cur ^ 1) * B_BUF);

    // All 8 fragments in distinct registers: ds_loads issue back-to-back and
    // drain with partial s_wait_dscnt under the WMMA burst.
    Frag a[4], b[4];
#pragma unroll
    for (int mi = 0; mi < 4; ++mi) {
      // A frag 16x32: lane row = l15, cols hi*16..+15 contiguous
      const __bf16* p = cA + (wm * 64 + mi * 16 + l15) * LDT + hi * 16;
      a[mi].u[0] = *(const uint4*)(p);
      a[mi].u[1] = *(const uint4*)(p + 8);
    }
#pragma unroll
    for (int ni = 0; ni < 4; ++ni) {
      // B frag 32x16: lane col = l15 (token), K = hi*16..+15 contiguous
      const __bf16* p = cB + (wn * 64 + ni * 16 + l15) * LDT + hi * 16;
      b[ni].u[0] = *(const uint4*)(p);
      b[ni].u[1] = *(const uint4*)(p + 8);
    }
#pragma unroll
    for (int ni = 0; ni < 4; ++ni)
#pragma unroll
      for (int mi = 0; mi < 4; ++mi)
        acc[mi][ni] = WMMA_BF16(a[mi].v, b[ni].v, acc[mi][ni]);
  }
}

// ----------------------------- stage 1: QKV GEMM ----------------------------
// qkv[o, tok] = sum_c W_qkv[o,c] * x[tok,c]; scatter to q/k/v layouts.
__global__ __launch_bounds__(256)
void gemm_qkv(const __bf16* __restrict__ W, const __bf16* __restrict__ X,
              __bf16* __restrict__ qbuf,   // [b][h][n][32]   (ReLU)
              __bf16* __restrict__ kbuf,   // [b][h][d][4096] (ReLU)
              __bf16* __restrict__ vbuf) { // [b][h][48][4096]
  __shared__ __bf16 sA[2 * A_BUF];
  __shared__ __bf16 sB[2 * B_BUF];
  v8f acc[4][4];
#pragma unroll
  for (int mi = 0; mi < 4; ++mi)
#pragma unroll
    for (int ni = 0; ni < 4; ++ni) acc[mi][ni] = v8f_zero();

  const int Mbase = blockIdx.y * 128;   // output channel o
  const int Nbase = blockIdx.x * 256;   // token
  gemm_mainloop(W + (size_t)Mbase * 1024, X + (size_t)Nbase * 1024, 1024, sA, sB, acc);

  const int lane = threadIdx.x & 31, wave = threadIdx.x >> 5;
  const int wm = wave >> 2, wn = wave & 3, l15 = lane & 15, hi = lane >> 4;
#pragma unroll
  for (int mi = 0; mi < 4; ++mi) {
    const int o0 = Mbase + wm * 64 + mi * 16 + hi * 8;  // 8 consecutive o
    const int which = o0 >> 10, rem = o0 & 1023;
    const int h = rem >> 5, d0 = rem & 31;
#pragma unroll
    for (int ni = 0; ni < 4; ++ni) {
      const int tok = Nbase + wn * 64 + ni * 16 + l15;
      const int b = tok >> 12, n = tok & 4095;
      v8f c = acc[mi][ni];
      if (which == 0) {            // q: ReLU, vectorized [b][h][n][d0..d0+7]
        Pack8 p;
#pragma unroll
        for (int i = 0; i < 8; ++i) p.h[i] = (__bf16)fmaxf(c[i], 0.0f);
        *(uint4*)(qbuf + (((size_t)(b * 32 + h) * 4096 + n) << 5) + d0) = p.u;
      } else if (which == 1) {     // k: ReLU, [b][h][d][n] (lane-coalesced)
#pragma unroll
        for (int i = 0; i < 8; ++i)
          kbuf[((size_t)(b * 32 + h) * 32 + d0 + i) * 4096 + n] = (__bf16)fmaxf(c[i], 0.0f);
      } else {                     // v: [b][h][d][n]
#pragma unroll
        for (int i = 0; i < 8; ++i)
          vbuf[((size_t)(b * 32 + h) * 48 + d0 + i) * 4096 + n] = (__bf16)c[i];
      }
    }
  }
}

// --------------------------- stage 2: attention core ------------------------
// Per (b,h): vk[48,32] = vpad[48,4096] * k^T ; out[d,n] = vk * q, / row 32.
__global__ __launch_bounds__(256)
void attn_core(const __bf16* __restrict__ qbuf, const __bf16* __restrict__ kbuf,
               const __bf16* __restrict__ vbuf, __bf16* __restrict__ tbuf) {
  __shared__ float  sPart[8][48 * 32];
  __shared__ __bf16 sVkb[48 * 32];

  const int bh = blockIdx.x;                 // 0..127
  const int b = bh >> 5, h = bh & 31;
  const int tid = threadIdx.x, lane = tid & 31, wave = tid >> 5;
  const int l15 = lane & 15, hi = lane >> 4;

  const __bf16* vg = vbuf + (size_t)bh * 48 * 4096;
  const __bf16* kg = kbuf + (size_t)bh * 32 * 4096;
  const __bf16* qg = qbuf + (size_t)bh * 4096 * 32;

  // ---- phase A: K-split vk accumulation (each wave owns 512 tokens) ----
  v8f acc[3][2];
#pragma unroll
  for (int mt = 0; mt < 3; ++mt)
#pragma unroll
    for (int et = 0; et < 2; ++et) acc[mt][et] = v8f_zero();

  const int n0 = wave * 512;
  for (int nn = n0; nn < n0 + 512; nn += 32) {
    Frag a[3], bf[2];
#pragma unroll
    for (int mt = 0; mt < 3; ++mt) {      // A = vpad rows (d), K = n
      const __bf16* p = vg + (size_t)(mt * 16 + l15) * 4096 + nn + hi * 16;
      a[mt].u[0] = *(const uint4*)(p);
      a[mt].u[1] = *(const uint4*)(p + 8);
    }
#pragma unroll
    for (int et = 0; et < 2; ++et) {      // B cols = e (k rows), K = n
      const __bf16* p = kg + (size_t)(et * 16 + l15) * 4096 + nn + hi * 16;
      bf[et].u[0] = *(const uint4*)(p);
      bf[et].u[1] = *(const uint4*)(p + 8);
    }
#pragma unroll
    for (int mt = 0; mt < 3; ++mt)
#pragma unroll
      for (int et = 0; et < 2; ++et)
        acc[mt][et] = WMMA_BF16(a[mt].v, bf[et].v, acc[mt][et]);
  }
  float* sp = sPart[wave];
#pragma unroll
  for (int mt = 0; mt < 3; ++mt)
#pragma unroll
    for (int et = 0; et < 2; ++et) {
      const int r = mt * 16 + hi * 8;
#pragma unroll
      for (int i = 0; i < 8; ++i)
        sp[(r + i) * 32 + et * 16 + l15] = acc[mt][et][i];
    }
  __syncthreads();

  // ---- cross-wave reduce + requantize vk to bf16 ----
  for (int e = tid * 6; e < tid * 6 + 6; ++e) {
    float s = 0.0f;
#pragma unroll
    for (int w = 0; w < 8; ++w) s += sPart[w][e];
    sVkb[e] = (__bf16)s;
  }
  __syncthreads();

  // ---- phase B: out = vk * q (single K=32 WMMA per tile), normalize ----
  Frag av[3];
#pragma unroll
  for (int mt = 0; mt < 3; ++mt) {
    const __bf16* p = sVkb + (mt * 16 + l15) * 32 + hi * 16;
    av[mt].u[0] = *(const uint4*)(p);
    av[mt].u[1] = *(const uint4*)(p + 8);
  }
  for (int c = 0; c < 32; ++c) {
    const int n = wave * 512 + c * 16 + l15;
    Frag bq;                               // B cols = tokens, K = e contiguous
    const __bf16* p = qg + (size_t)n * 32 + hi * 16;
    bq.u[0] = *(const uint4*)(p);
    bq.u[1] = *(const uint4*)(p + 8);

    v8f o0 = WMMA_BF16(av[0].v, bq.v, v8f_zero());
    v8f o1 = WMMA_BF16(av[1].v, bq.v, v8f_zero());
    v8f o2 = WMMA_BF16(av[2].v, bq.v, v8f_zero());

    // denominator = row 32 (o2 vgpr0, lanes 0..15); broadcast to upper half
    float den = __int_as_float(
        __builtin_amdgcn_ds_bpermute(l15 << 2, __float_as_int(o2[0])));
    const float rn = 1.0f / (den + 1e-15f);

#pragma unroll
    for (int m = 0; m < 2; ++m) {
      v8f om = (m == 0) ? o0 : o1;
      Pack8 pk;
#pragma unroll
      for (int i = 0; i < 8; ++i) pk.h[i] = (__bf16)(om[i] * rn);
      const int d0b = m * 16 + hi * 8;
      __bf16* dst = tbuf + ((size_t)(b * 4096 + n)) * 1024 + h * 32 + d0b;
      *(uint4*)dst = pk.u;
    }
  }
}

// --------------------------- stage 3: output proj ---------------------------
__global__ __launch_bounds__(256)
void gemm_proj(const __bf16* __restrict__ Wp, const __bf16* __restrict__ T,
               const float* __restrict__ bias, float* __restrict__ out) {
  __shared__ __bf16 sA[2 * A_BUF];
  __shared__ __bf16 sB[2 * B_BUF];
  v8f acc[4][4];
#pragma unroll
  for (int mi = 0; mi < 4; ++mi)
#pragma unroll
    for (int ni = 0; ni < 4; ++ni) acc[mi][ni] = v8f_zero();

  const int Mbase = blockIdx.y * 128;   // output channel o (0..1023)
  const int Nbase = blockIdx.x * 256;   // token
  gemm_mainloop(Wp + (size_t)Mbase * 1024, T + (size_t)Nbase * 1024, 1024, sA, sB, acc);

  const int lane = threadIdx.x & 31, wave = threadIdx.x >> 5;
  const int wm = wave >> 2, wn = wave & 3, l15 = lane & 15, hi = lane >> 4;
#pragma unroll
  for (int mi = 0; mi < 4; ++mi) {
    const int o0 = Mbase + wm * 64 + mi * 16 + hi * 8;
    const float4 bv0 = *(const float4*)(bias + o0);
    const float4 bv1 = *(const float4*)(bias + o0 + 4);
#pragma unroll
    for (int ni = 0; ni < 4; ++ni) {
      const int tok = Nbase + wn * 64 + ni * 16 + l15;
      v8f c = acc[mi][ni];
      float* dst = out + (size_t)tok * 1024 + o0;
      float4 r0 = {c[0] + bv0.x, c[1] + bv0.y, c[2] + bv0.z, c[3] + bv0.w};
      float4 r1 = {c[4] + bv1.x, c[5] + bv1.y, c[6] + bv1.z, c[7] + bv1.w};
      *(float4*)(dst)     = r0;
      *(float4*)(dst + 4) = r1;
    }
  }
}

// ------------------------------- launcher -----------------------------------
extern "C" void kernel_launch(void* const* d_in, const int* in_sizes, int n_in,
                              void* d_out, int out_size, void* d_ws, size_t ws_size,
                              hipStream_t stream) {
  const float* x     = (const float*)d_in[0];   // (4,4096,1024)
  const float* Wqkv  = (const float*)d_in[1];   // (3072,1024)
  const float* Wproj = (const float*)d_in[2];   // (1024,1024)
  const float* bproj = (const float*)d_in[3];   // (1024,)

  char* ws = (char*)d_ws;
  __bf16* xb   = (__bf16*)(ws);                 // 33,554,432 B
  __bf16* wqb  = (__bf16*)(ws + 33554432);      //  6,291,456 B
  __bf16* wpb  = (__bf16*)(ws + 39845888);      //  2,097,152 B
  __bf16* qbuf = (__bf16*)(ws + 41943040);      // 33,554,432 B  [b][h][n][32]
  __bf16* kbuf = (__bf16*)(ws + 75497472);      // 33,554,432 B  [b][h][32][n]
  __bf16* vbuf = (__bf16*)(ws + 109051904);     // 50,331,648 B  [b][h][48][n]
  __bf16* tbuf = (__bf16*)(ws + 159383552);     // 33,554,432 B  [b][n][1024]
  float* out = (float*)d_out;

  cvt_f32_bf16<<<16777216 / 1024, 256, 0, stream>>>(x, xb, 16777216);
  cvt_f32_bf16<<<3145728 / 1024, 256, 0, stream>>>(Wqkv, wqb, 3145728);
  cvt_f32_bf16<<<1048576 / 1024, 256, 0, stream>>>(Wproj, wpb, 1048576);
  init_vpad<<<8388608 / 256, 256, 0, stream>>>(vbuf);

  gemm_qkv<<<dim3(64, 24), 256, 0, stream>>>(wqb, xb, qbuf, kbuf, vbuf);
  attn_core<<<128, 256, 0, stream>>>(qbuf, kbuf, vbuf, tbuf);
  gemm_proj<<<dim3(64, 8), 256, 0, stream>>>(wpb, tbuf, bproj, out);
}
